// TrajPredictor_78185584656913
// MI455X (gfx1250) — compile-verified, matrix operand order checked by reading, not run on hardware
//
#include <hip/hip_runtime.h>
#include <hip/hip_bf16.h>
#include <math.h>

// ---- problem constants (from setup_inputs) ----
#define HDIM   64
#define GDIM   192        // 3*H
#define SLEN   100
#define BDIM   128
#define ADIM   32
#define INDIM  3
#define NROW   (BDIM*ADIM)   // 4096 independent sequences
#define PRED   12
#define DECK   131           // IN + 2H decoder input width

typedef __attribute__((ext_vector_type(16))) _Float16 v16h;
typedef __attribute__((ext_vector_type(8)))  float    v8f;

// ---------------- WMMA fragment helpers (CDNA5 16x16x32 f16 layouts) ----------------
// A (16x32 f16): lane L -> row m=L&15; half i in [0,16): k = (i>=8?16:0) + ((L>=16)?8:0) + (i&7)
// B (32x16 f16): lane L -> col n=L&15; same k map.
// C/D (16x16 f32, 8 VGPRs): lane L -> col n=L&15, vgpr v -> row m = v + 8*(L>=16)

__device__ __forceinline__ int kmap(int lane, int i) {
    return ((i >= 8) ? 16 : 0) + ((lane & 16) >> 1) + (i & 7);
}

__device__ __forceinline__ v8f vzero8() {
    v8f a;
#pragma unroll
    for (int i = 0; i < 8; ++i) a[i] = 0.f;
    return a;
}

__device__ __forceinline__ v16h hzero16() {
    v16h a;
#pragma unroll
    for (int i = 0; i < 16; ++i) a[i] = (_Float16)0.f;
    return a;
}

__device__ __forceinline__ v8f wmma32(v16h a, v16h b, v8f c) {
    // (neg_a, A, neg_b, B, c_mod, C, reuse_a, reuse_b)
    return __builtin_amdgcn_wmma_f32_16x16x32_f16(false, a, false, b, (short)0, c, false, false);
}

// B fragment gathered from global f32 weights, row-major W[n][k] (ldk); used ONLY by
// the one-shot pack kernel.
__device__ __forceinline__ v16h makeB(const float* __restrict__ W, int ldk, int kmax,
                                      int ncol, int kc, int lane) {
    v16h b;
    int n = ncol + (lane & 15);
#pragma unroll
    for (int i = 0; i < 16; ++i) {
        int k = kc * 32 + kmap(lane, i);
        b[i] = (_Float16)((k < kmax) ? W[n * ldk + k] : 0.f);
    }
    return b;
}

// A fragment from f32 LDS tile (rows of this wave), zero-pad k>=kmax
__device__ __forceinline__ v16h makeA_f32(const float* base, int stride, int kc, int kmax, int lane) {
    v16h a;
    int m = lane & 15;
#pragma unroll
    for (int i = 0; i < 16; ++i) {
        int k = kc * 32 + kmap(lane, i);
        a[i] = (k < kmax) ? (_Float16)base[m * stride + k] : (_Float16)0.f;
    }
    return a;
}

// A fragment from f16 LDS tile
__device__ __forceinline__ v16h makeA_f16(const _Float16* base, int stride, int kc, int lane) {
    v16h a;
    int m = lane & 15;
#pragma unroll
    for (int i = 0; i < 16; ++i) {
        int k = kc * 32 + kmap(lane, i);
        a[i] = base[m * stride + k];
    }
    return a;
}

// Coalesced B-fragment load from the packed-f16 global buffer (2x global_load_b128/lane)
__device__ __forceinline__ v16h loadFrag(const uint4* __restrict__ pack, int frag, int lane) {
    union { v16h v; uint4 q[2]; } u;
    int s = (frag * 32 + lane) * 2;
    u.q[0] = pack[s]; u.q[1] = pack[s + 1];
    return u.v;
}

__device__ __forceinline__ float sigmoidf_(float x) { return 1.f / (1.f + __expf(-x)); }

// ---- packed fragment buffer layout (frag index bases); frag = 64 uint4 = 32 lanes * 32B
#define PK_ENCF 0      // enc fwd: 24 Whh frags (kc*12+nt) then 12 Wih frags  -> 36
#define PK_ENCB 36     // enc bwd: same                                       -> 36
#define PK_ATT1 72     // attn_W[:, :64]:  kc*4+nt, kc<2                      ->  8
#define PK_ATT2 80     // attn_W[:,64:192]: kc*4+nt, kc<4                     -> 16
#define PK_DECH 96     // dec_Whh: kc*12+nt, kc<2                             -> 24
#define PK_DECI 120    // dec_Wih: kc*12+nt, kc<5 (K=131 padded to 160)       -> 60
#define PK_NFRAG 180

// =====================================================================================
// Kernel 0: one-shot weight pre-swizzle into fragment-major f16 (180 KB, L2-resident).
// =====================================================================================
__global__ __launch_bounds__(256) void pack_weights(
    const float* __restrict__ Whh_f, const float* __restrict__ Wih_f,
    const float* __restrict__ Whh_b, const float* __restrict__ Wih_b,
    const float* __restrict__ attn_W,
    const float* __restrict__ dWhh, const float* __restrict__ dWih,
    uint4* __restrict__ pack)
{
    int slot = blockIdx.x * blockDim.x + threadIdx.x;   // PK_NFRAG*32 slots
    if (slot >= PK_NFRAG * 32) return;
    int f = slot >> 5, L = slot & 31;
    v16h v;
    if      (f < 24)  { int g = f;       v = makeB(Whh_f, 64, 64, (g % 12) * 16, g / 12, L); }
    else if (f < 36)  { int g = f - 24;  v = makeB(Wih_f, 3, 3, g * 16, 0, L); }
    else if (f < 60)  { int g = f - 36;  v = makeB(Whh_b, 64, 64, (g % 12) * 16, g / 12, L); }
    else if (f < 72)  { int g = f - 60;  v = makeB(Wih_b, 3, 3, g * 16, 0, L); }
    else if (f < 80)  { int g = f - 72;  v = makeB(attn_W, GDIM, 64, (g % 4) * 16, g / 4, L); }
    else if (f < 96)  { int g = f - 80;  v = makeB(attn_W + 64, GDIM, 128, (g % 4) * 16, g / 4, L); }
    else if (f < 120) { int g = f - 96;  v = makeB(dWhh, 64, 64, (g % 12) * 16, g / 12, L); }
    else              { int g = f - 120; v = makeB(dWih, DECK, DECK, (g % 12) * 16, g / 12, L); }
    union { v16h v; uint4 q[2]; } u; u.v = v;
    pack[slot * 2] = u.q[0]; pack[slot * 2 + 1] = u.q[1];
}

// =====================================================================================
// Kernel 1: encoder GRU (one direction per launch). 32 rows / WG (2 waves, 64 thr),
// grid = 4096/32 = 128 WGs. h kept in LDS f32; packed B-fragments copied coalesced
// into LDS once (amortized over 100 steps). 36 v_wmma per wave per step.
// =====================================================================================
#define EROWS 32
#define ETHREADS 64
#define ENC_SMEM (36*32*32 + EROWS*64*4 + 2*192*4)

__global__ __launch_bounds__(ETHREADS) void enc_gru(
    const float* __restrict__ src, const uint4* __restrict__ pack,
    const float* __restrict__ bih, const float* __restrict__ bhh,
    _Float16* __restrict__ eo, float* __restrict__ hfin, int dir)
{
    extern __shared__ char smem[];
    uint4* bpack  = (uint4*)smem;                 // 36 frags (24 Whh + 12 Wih) * 64 uint4
    float* hbuf   = (float*)(bpack + 36 * 64);    // [32][64]
    float* bI     = hbuf + EROWS * 64;            // 192
    float* bH     = bI + 192;                     // 192

    const int tid = threadIdx.x;
    const int wave = tid >> 5, lane = tid & 31, lm = lane & 15, lhi = lane >> 4;

    for (int i = tid; i < EROWS * 64; i += ETHREADS) hbuf[i] = 0.f;
    for (int i = tid; i < 192; i += ETHREADS) { bI[i] = bih[i]; bH[i] = bhh[i]; }
    // coalesced copy of this direction's 36 pre-swizzled fragments into LDS
    {
        const uint4* gp = pack + (size_t)(dir ? PK_ENCB : PK_ENCF) * 64;
        for (int i = tid; i < 36 * 64; i += ETHREADS) bpack[i] = gp[i];
    }
    __syncthreads();

    const int row0 = blockIdx.x * EROWS + wave * 16;
    const float* hrow = hbuf + (wave * 16) * 64;

    for (int t = 0; t < SLEN; ++t) {
        int tt = dir ? (SLEN - 1 - t) : t;
        v16h a0 = makeA_f32(hrow, 64, 0, 64, lane);
        v16h a1 = makeA_f32(hrow, 64, 1, 64, lane);
        v16h ax = hzero16();
        if (lhi == 0) {   // k=0..2 live in lo-lane halves 0..2
            int n = row0 + lm; int b = n >> 5, aa = n & 31;
            const float* xp = src + ((size_t)b * SLEN + tt) * (ADIM * INDIM) + aa * INDIM;
            ax[0] = (_Float16)xp[0]; ax[1] = (_Float16)xp[1]; ax[2] = (_Float16)xp[2];
        }

        v8f crz[8], chn[4], cin[4];
#pragma unroll
        for (int nt = 0; nt < 12; ++nt) {
            union { v16h v; uint4 q[2]; } u0, u1, ux;
            int s0 = (nt * 32 + lane) * 2;            // Whh kc=0
            int s1 = ((12 + nt) * 32 + lane) * 2;     // Whh kc=1
            int sx = ((24 + nt) * 32 + lane) * 2;     // Wih
            u0.q[0] = bpack[s0]; u0.q[1] = bpack[s0 + 1];
            u1.q[0] = bpack[s1]; u1.q[1] = bpack[s1 + 1];
            ux.q[0] = bpack[sx]; ux.q[1] = bpack[sx + 1];
            v8f hh = vzero8();
            hh = wmma32(a0, u0.v, hh);
            hh = wmma32(a1, u1.v, hh);
            if (nt < 8) {            // r,z gates: gi+gh merged
                crz[nt] = wmma32(ax, ux.v, hh);
            } else {                 // n gate: keep inn / hn separate
                chn[nt - 8] = hh;
                cin[nt - 8] = wmma32(ax, ux.v, vzero8());
            }
        }
        // GRU epilogue (rows partitioned per wave -> no barrier needed)
#pragma unroll
        for (int q = 0; q < 4; ++q) {
#pragma unroll
            for (int v = 0; v < 8; ++v) {
                int m = wave * 16 + v + 8 * lhi;
                int j = q * 16 + lm;
                float r  = sigmoidf_(crz[q][v]     + bI[j]       + bH[j]);
                float z  = sigmoidf_(crz[4 + q][v] + bI[64 + j]  + bH[64 + j]);
                float nn = tanhf(cin[q][v] + bI[128 + j] + r * (chn[q][v] + bH[128 + j]));
                float hold = hbuf[m * 64 + j];
                float hnew = (1.f - z) * nn + z * hold;
                hbuf[m * 64 + j] = hnew;
                int mg = blockIdx.x * EROWS + m;
                eo[((size_t)mg * SLEN + tt) * 128 + dir * 64 + j] = (_Float16)hnew;
            }
        }
    }
    __syncthreads();
    for (int i = tid; i < EROWS * 64; i += ETHREADS)
        hfin[(size_t)blockIdx.x * EROWS * 64 + i] = hbuf[i];
}

// =====================================================================================
// Kernel 2: h_dec0 = [hf|hb] @ e2d_W.T + e2d_b ; dec_in0 = x[:,-1,:]
// =====================================================================================
__global__ __launch_bounds__(256) void e2d_init(
    const float* __restrict__ hf, const float* __restrict__ hb,
    const float* __restrict__ e2dW, const float* __restrict__ e2db,
    const float* __restrict__ src, float* __restrict__ hdec, float* __restrict__ dinp)
{
    int idx = blockIdx.x * blockDim.x + threadIdx.x;   // N*64 threads
    int n = idx >> 6, j = idx & 63;
    float s = e2db[j];
    const float* wr = e2dW + j * 128;
#pragma unroll 4
    for (int k = 0; k < 64; ++k) s += hf[n * 64 + k] * wr[k];
#pragma unroll 4
    for (int k = 0; k < 64; ++k) s += hb[n * 64 + k] * wr[64 + k];
    hdec[idx] = s;
    if (j < INDIM) {
        int b = n >> 5, a = n & 31;
        dinp[n * INDIM + j] = src[((size_t)b * SLEN + (SLEN - 1)) * (ADIM * INDIM) + a * INDIM + j];
    }
}

// =====================================================================================
// Kernel 3: step-invariant attention part:
//   ep[n,s,j] = enc_out[n,s,:] @ attn_W[j,64:192] + attn_b[j]
// GEMM [N*S,128]x[128,64] via WMMA. 128 rows / WG (8 waves, 256 thr), grid 3200.
// =====================================================================================
#define EPROWS 128
__global__ __launch_bounds__(256) void attn_pre(
    const _Float16* __restrict__ eo, const uint4* __restrict__ pack,
    const float* __restrict__ attn_b, float* __restrict__ ep)
{
    extern __shared__ char smem[];
    _Float16* atile = (_Float16*)smem;   // [128][128] f16 = 32KB
    const int tid = threadIdx.x, wave = tid >> 5, lane = tid & 31, lm = lane & 15, lhi = lane >> 4;
    const size_t rowbase = (size_t)blockIdx.x * EPROWS;

    const uint4* s4 = (const uint4*)(eo + rowbase * 128);
    uint4* d4 = (uint4*)atile;
    for (int i = tid; i < EPROWS * 128 / 8; i += 256) d4[i] = s4[i];
    __syncthreads();

    v16h a[4];
#pragma unroll
    for (int kc = 0; kc < 4; ++kc) a[kc] = makeA_f16(atile + wave * 16 * 128, 128, kc, lane);

#pragma unroll
    for (int nt = 0; nt < 4; ++nt) {
        v8f acc = vzero8();
#pragma unroll
        for (int kc = 0; kc < 4; ++kc)
            acc = wmma32(a[kc], loadFrag(pack, PK_ATT2 + kc * 4 + nt, lane), acc);
#pragma unroll
        for (int v = 0; v < 8; ++v) {
            int m = wave * 16 + v + 8 * lhi;
            int j = nt * 16 + lm;
            ep[(rowbase + m) * 64 + j] = acc[v] + attn_b[j];
        }
    }
}

// =====================================================================================
// Kernel 4: one decoder step. 32 rows / WG (2 waves, 64 thr), grid 128. Launched 12x.
//  A: h_part = h @ attn_W[:,:64].T           (WMMA, packed-frag B)
//  B: score/softmax/context                  (VALU; ep f32 + enc_out f16 from L2)
//  C: decoder GRU, gi K=131 (5 k-chunks)     (WMMA, packed-frag B)
//  D: pred = h @ out_W.T + out_b -> output & next input
// =====================================================================================
#define DROWS 32
#define DTHREADS 64
#define DEC_SMEM (DROWS*64*4 + DROWS*64*4 + DROWS*SLEN*4 + DROWS*132*4 + 64*4 + 2*192*4)

__global__ __launch_bounds__(DTHREADS) void dec_step(
    const uint4* __restrict__ pack, const float* __restrict__ attn_v,
    const float* __restrict__ dbih, const float* __restrict__ dbhh,
    const float* __restrict__ outW, const float* __restrict__ outb,
    const _Float16* __restrict__ eo, const float* __restrict__ ep,
    float* __restrict__ hstate, float* __restrict__ dinp,
    float* __restrict__ out, int step)
{
    extern __shared__ char smem[];
    float* hbuf  = (float*)smem;                  // [32][64]
    float* hpart = hbuf + DROWS * 64;             // [32][64]
    float* sc    = hpart + DROWS * 64;            // [32][100]
    float* rin   = sc + DROWS * SLEN;             // [32][132]  (inp(3)|context(128)|pad)
    float* av    = rin + DROWS * 132;             // [64] attn_v
    float* bI    = av + 64;                       // [192] dec_bih
    float* bH    = bI + 192;                      // [192] dec_bhh

    const int tid = threadIdx.x, wave = tid >> 5, lane = tid & 31, lm = lane & 15, lhi = lane >> 4;
    const size_t row0 = (size_t)blockIdx.x * DROWS;

    for (int i = tid; i < DROWS * 64; i += DTHREADS) hbuf[i] = hstate[row0 * 64 + i];
    for (int i = tid; i < 64; i += DTHREADS) av[i] = attn_v[i];
    for (int i = tid; i < 192; i += DTHREADS) { bI[i] = dbih[i]; bH[i] = dbhh[i]; }
    __syncthreads();

    // ---- phase A: h_part = h @ attn_W[:, :64].T ----
    {
        const float* hrow = hbuf + wave * 16 * 64;
        v16h a0 = makeA_f32(hrow, 64, 0, 64, lane);
        v16h a1 = makeA_f32(hrow, 64, 1, 64, lane);
#pragma unroll
        for (int nt = 0; nt < 4; ++nt) {
            v8f acc = vzero8();
            acc = wmma32(a0, loadFrag(pack, PK_ATT1 + 0 * 4 + nt, lane), acc);
            acc = wmma32(a1, loadFrag(pack, PK_ATT1 + 1 * 4 + nt, lane), acc);
#pragma unroll
            for (int v = 0; v < 8; ++v) {
                int m = wave * 16 + v + 8 * lhi;
                hpart[m * 64 + nt * 16 + lm] = acc[v];
            }
        }
    }
    __syncthreads();

    // ---- phase B: scores, softmax, context ----
    {
        int r = tid >> 1, half = tid & 1;      // 2 threads per row, 50 s each
        for (int s = half * 50; s < half * 50 + 50; ++s) {
            const float* e = ep + ((row0 + r) * SLEN + s) * 64;
            float acc = 0.f;
#pragma unroll 4
            for (int j = 0; j < 64; ++j) acc += av[j] * tanhf(hpart[r * 64 + j] + e[j]);
            sc[r * SLEN + s] = acc;
        }
    }
    __syncthreads();
    if (tid < DROWS) {
        int r = tid; float mx = -1e30f;
        for (int s = 0; s < SLEN; ++s) mx = fmaxf(mx, sc[r * SLEN + s]);
        float sum = 0.f;
        for (int s = 0; s < SLEN; ++s) { float e = __expf(sc[r * SLEN + s] - mx); sc[r * SLEN + s] = e; sum += e; }
        float inv = 1.f / sum;
        for (int s = 0; s < SLEN; ++s) sc[r * SLEN + s] *= inv;
    }
    __syncthreads();
    for (int idx = tid; idx < DROWS * 128; idx += DTHREADS) {
        int r = idx >> 7, k = idx & 127;
        const _Float16* er = eo + ((row0 + r) * SLEN) * 128 + k;
        float acc = 0.f;
#pragma unroll 4
        for (int s = 0; s < SLEN; ++s) acc += sc[r * SLEN + s] * (float)er[s * 128];
        rin[r * 132 + 3 + k] = acc;
    }
    for (int idx = tid; idx < DROWS * INDIM; idx += DTHREADS) {
        int r = idx / 3, c = idx % 3;
        rin[r * 132 + c] = dinp[(row0 + r) * INDIM + c];
    }
    __syncthreads();

    // ---- phase C: decoder GRU (gh: K=64, gi: K=131 -> 5 chunks) ----
    {
        const float* hrow = hbuf + wave * 16 * 64;
        const float* rrow = rin + wave * 16 * 132;
        v16h ah0 = makeA_f32(hrow, 64, 0, 64, lane);
        v16h ah1 = makeA_f32(hrow, 64, 1, 64, lane);
        v16h axf[5];
#pragma unroll
        for (int kc = 0; kc < 5; ++kc) axf[kc] = makeA_f32(rrow, 132, kc, DECK, lane);

        v8f crz[8], chn[4], cin[4];
#pragma unroll
        for (int nt = 0; nt < 12; ++nt) {
            v8f hh = vzero8();
            hh = wmma32(ah0, loadFrag(pack, PK_DECH + 0 * 12 + nt, lane), hh);
            hh = wmma32(ah1, loadFrag(pack, PK_DECH + 1 * 12 + nt, lane), hh);
            v8f xx = (nt < 8) ? hh : vzero8();
#pragma unroll
            for (int kc = 0; kc < 5; ++kc)
                xx = wmma32(axf[kc], loadFrag(pack, PK_DECI + kc * 12 + nt, lane), xx);
            if (nt < 8) crz[nt] = xx;
            else { chn[nt - 8] = hh; cin[nt - 8] = xx; }
        }
#pragma unroll
        for (int q = 0; q < 4; ++q) {
#pragma unroll
            for (int v = 0; v < 8; ++v) {
                int m = wave * 16 + v + 8 * lhi;
                int j = q * 16 + lm;
                float r_ = sigmoidf_(crz[q][v]     + bI[j]      + bH[j]);
                float z_ = sigmoidf_(crz[4 + q][v] + bI[64 + j] + bH[64 + j]);
                float nn = tanhf(cin[q][v] + bI[128 + j] + r_ * (chn[q][v] + bH[128 + j]));
                float hold = hbuf[m * 64 + j];
                float hnew = (1.f - z_) * nn + z_ * hold;
                hbuf[m * 64 + j] = hnew;
                hstate[(row0 + m) * 64 + j] = hnew;
            }
        }
    }
    __syncthreads();

    // ---- phase D: output projection ----
    for (int idx = tid; idx < DROWS * INDIM; idx += DTHREADS) {
        int r = idx / 3, c = idx % 3;
        float s = outb[c];
        const float* w = outW + c * 64;
#pragma unroll 4
        for (int j = 0; j < 64; ++j) s += hbuf[r * 64 + j] * w[j];
        size_t n = row0 + r; int b = (int)(n >> 5), a = (int)(n & 31);
        out[((size_t)b * PRED + step) * (ADIM * INDIM) + a * INDIM + c] = s;
        dinp[n * INDIM + c] = s;
    }
}

// =====================================================================================
extern "C" void kernel_launch(void* const* d_in, const int* in_sizes, int n_in,
                              void* d_out, int out_size, void* d_ws, size_t ws_size,
                              hipStream_t stream)
{
    (void)in_sizes; (void)n_in; (void)out_size; (void)ws_size;
    const float* src    = (const float*)d_in[0];
    const float* Wih_f  = (const float*)d_in[1];
    const float* Whh_f  = (const float*)d_in[2];
    const float* bih_f  = (const float*)d_in[3];
    const float* bhh_f  = (const float*)d_in[4];
    const float* Wih_b  = (const float*)d_in[5];
    const float* Whh_b  = (const float*)d_in[6];
    const float* bih_b  = (const float*)d_in[7];
    const float* bhh_b  = (const float*)d_in[8];
    const float* attn_W = (const float*)d_in[9];
    const float* attn_b = (const float*)d_in[10];
    const float* attn_v = (const float*)d_in[11];
    const float* e2dW   = (const float*)d_in[12];
    const float* e2db   = (const float*)d_in[13];
    const float* dWih   = (const float*)d_in[14];
    const float* dWhh   = (const float*)d_in[15];
    const float* dbih   = (const float*)d_in[16];
    const float* dbhh   = (const float*)d_in[17];
    const float* outW   = (const float*)d_in[18];
    const float* outb   = (const float*)d_in[19];
    // d_in[20] = pred_len (==12 in setup) — hardcoded PRED

    // workspace carve (f16 enc_out = 105MB -> fits 192MB L2 for decoder reuse)
    char* ws = (char*)d_ws;
    size_t off = 0;
    _Float16* eo = (_Float16*)(ws + off); off += (size_t)NROW * SLEN * 128 * sizeof(_Float16);
    float* ep    = (float*)(ws + off);    off += (size_t)NROW * SLEN * 64 * sizeof(float);
    float* hf    = (float*)(ws + off);    off += (size_t)NROW * 64 * sizeof(float);
    float* hb    = (float*)(ws + off);    off += (size_t)NROW * 64 * sizeof(float);
    float* hdec  = (float*)(ws + off);    off += (size_t)NROW * 64 * sizeof(float);
    float* dinp  = (float*)(ws + off);    off += (size_t)NROW * INDIM * sizeof(float);
    uint4* pack  = (uint4*)(ws + off);    off += (size_t)PK_NFRAG * 32 * 32;   // 180KB packed frags

    // one-shot weight pre-swizzle into WMMA fragment layout (f16)
    pack_weights<<<(PK_NFRAG * 32 + 255) / 256, 256, 0, stream>>>(
        Whh_f, Wih_f, Whh_b, Wih_b, attn_W, dWhh, dWih, pack);

    // encoder, both directions
    enc_gru<<<NROW / EROWS, ETHREADS, ENC_SMEM, stream>>>(src, pack, bih_f, bhh_f, eo, hf, 0);
    enc_gru<<<NROW / EROWS, ETHREADS, ENC_SMEM, stream>>>(src, pack, bih_b, bhh_b, eo, hb, 1);

    // decoder init + step-invariant attention GEMM
    e2d_init<<<(NROW * 64) / 256, 256, 0, stream>>>(hf, hb, e2dW, e2db, src, hdec, dinp);
    attn_pre<<<(NROW * SLEN) / EPROWS, 256, EPROWS * 128 * 2, stream>>>(eo, pack, attn_b, ep);

    // 12 sequential decoder steps
    for (int p = 0; p < PRED; ++p) {
        dec_step<<<NROW / DROWS, DTHREADS, DEC_SMEM, stream>>>(
            pack, attn_v, dbih, dbhh, outW, outb,
            eo, ep, hdec, dinp, (float*)d_out, p);
    }
}
